// NeuralCDE_18098992185902
// MI455X (gfx1250) — matrix operand order repaired and checked by reading, not compiled, not run on hardware
//
#include <hip/hip_runtime.h>
#include <hip/hip_bf16.h>

// Neural CDE forward on gfx1250 (MI455X).
// B=128, T=1024, D=8, H=64, W=128.  8 persistent workgroups x 256 threads,
// each owns 16 batch rows (one WMMA M-tile) and scans all 1023 RK4 steps.
// All weights converted to bf16 and held in LDS (~240KB of 320KB/WGP).
// Per-step spline coefficients staged via GLOBAL_LOAD_ASYNC_TO_LDS (ASYNCcnt).

typedef __attribute__((ext_vector_type(16))) __bf16 v16bf;
typedef __attribute__((ext_vector_type(8)))  float  v8f;

#define Tm1 1023
#define Dn  8
#define Hn  64
#define Wn  128
#define HDn 512
#define MB  16

__device__ __forceinline__ unsigned short f2bf(float f) {
  unsigned u = __float_as_uint(f);
  u += 0x7FFFu + ((u >> 16) & 1u);          // round-to-nearest-even
  return (unsigned short)(u >> 16);
}
__device__ __forceinline__ float softplus_f(float x) {
  return fmaxf(x, 0.f) + __logf(1.f + __expf(-fabsf(x)));
}
__device__ __forceinline__ float tanh_f(float x) {
  float e = __expf(-2.f * fabsf(x));
  float t = (1.f - e) / (1.f + e);
  return copysignf(t, x);
}

// One 16x16 output tile of  A[16,K](bf16, row-major LDS)  x  B[K, nt*16..+16]
// using v_wmma_f32_16x16x32_bf16.  B is pre-packed in fragment order:
// uint index ((nt*KB+kb)*256 + lane*8 + v) holds N=lane&15 at K=kb*32+(v>>2)*16+(lane>>4)*8+(v&3)*2 (+0,+1).
template <int K>
__device__ __forceinline__ v8f wave_gemm_tile(const unsigned short* __restrict__ A,
                                              const unsigned* __restrict__ Bw,
                                              int nt, int lane, v8f acc) {
  constexpr int KB = K >> 5;
  const int m    = lane & 15;
  const int half = lane >> 4;
  const unsigned short* arow = A + m * K + half * 8;
#pragma unroll
  for (int kb = 0; kb < KB; ++kb) {
    union { unsigned u[8]; v16bf v; } af, bfr;
#pragma unroll
    for (int v = 0; v < 8; ++v)   // ISA 16-bit A 16x32 layout: k=(v>>2)*16+half*8+(v&3)*2
      af.u[v] = *(const unsigned*)(arow + kb * 32 + ((v >> 2) * 16 + (v & 3) * 2));
    const unsigned* bb = Bw + (nt * KB + kb) * 256 + lane * 8;
#pragma unroll
    for (int v = 0; v < 8; ++v) bfr.u[v] = bb[v];
    acc = __builtin_amdgcn_wmma_f32_16x16x32_bf16(false, af.v, false, bfr.v,
                                                  (short)0, acc, false, false);
  }
  return acc;
}

// Convert f32 weight W[N][K] (row-major, global) into bf16 fragment-packed LDS image.
__device__ void pack_weight(const float* __restrict__ Wg, unsigned* __restrict__ Wl,
                            int K, int total_uints, int kbits, int tid) {
  const int KB = K >> 5;
  for (int i = tid; i < total_uints; i += 256) {
    int v  = i & 7;
    int L  = (i >> 3) & 31;
    int c  = i >> 8;
    int kb = c & (KB - 1);
    int nt = c >> kbits;
    int k  = kb * 32 + (v >> 2) * 16 + (L >> 4) * 8 + (v & 3) * 2;
    int n0 = nt * 16 + (L & 15);
    unsigned lo = f2bf(Wg[(size_t)n0 * K + k]);
    unsigned hi = f2bf(Wg[(size_t)n0 * K + k + 1]);
    Wl[i] = lo | (hi << 16);
  }
}

__global__ void __launch_bounds__(256, 1)
ncde_kernel(const float* __restrict__ ts,
            const float* __restrict__ cd, const float* __restrict__ cc,
            const float* __restrict__ cb, const float* __restrict__ ca,
            const float* __restrict__ iw0, const float* __restrict__ ib0,
            const float* __restrict__ iw1, const float* __restrict__ ib1,
            const float* __restrict__ iw2, const float* __restrict__ ib2,
            const float* __restrict__ fw0, const float* __restrict__ fb0,
            const float* __restrict__ fw1, const float* __restrict__ fb1,
            const float* __restrict__ fw2, const float* __restrict__ fb2,
            const float* __restrict__ lw,  const float* __restrict__ lb,
            float* __restrict__ out) {
  extern __shared__ char smem[];
  unsigned* fw0u = (unsigned*)smem;          // 4096 uints  (128x64  bf16)
  unsigned* fw1u = fw0u + 4096;              // 8192 uints  (128x128 bf16)
  unsigned* fw2u = fw1u + 8192;              // 32768 uints (512x128 bf16)
  float* fb0s = (float*)(fw2u + 32768);      // 128
  float* fb1s = fb0s + Wn;                   // 128
  float* fb2s = fb1s + Wn;                   // 512
  float* yf   = fb2s + HDn;                  // 16x64 f32 state
  float* ksum = yf + MB * Hn;                // 16x64 RK accumulator
  float* kst  = ksum + MB * Hn;              // 16x64 stage k
  float* Fb   = kst + MB * Hn;               // 16x512 f32 (tanh output / init scratch)
  float* dCs  = Fb + MB * HDn;               // 16x8
  float* cCs  = dCs + 128;
  float* bCs  = cCs + 128;
  float* dxs  = bCs + 128;
  unsigned short* ysb = (unsigned short*)(dxs + 128); // 16x64  bf16 stage input
  unsigned short* h1b = ysb + MB * Hn;                // 16x128 bf16
  unsigned short* h2b = h1b + MB * Wn;                // 16x128 bf16

  const int tid  = threadIdx.x;
  const int lane = tid & 31;
  const int wave = tid >> 5;
  const int b0   = blockIdx.x * MB;

  // ---- prologue: stage weights into LDS (bf16, fragment-packed) ----
  pack_weight(fw0, fw0u, Hn, 4096, 1, tid);
  pack_weight(fw1, fw1u, Wn, 8192, 2, tid);
  pack_weight(fw2, fw2u, Wn, 32768, 2, tid);
  for (int i = tid; i < Wn; i += 256) { fb0s[i] = fb0[i]; fb1s[i] = fb1[i]; }
  for (int i = tid; i < HDn; i += 256) fb2s[i] = fb2[i];
  __syncthreads();

  // ---- y0 = init MLP on x = coeff_a[:,0,:]  (done once, plain VALU) ----
  for (int i = tid; i < MB * Wn; i += 256) {           // h0 = relu(x @ iw0.T + ib0)
    int m = i >> 7, w = i & 127;
    float a = ib0[w];
#pragma unroll
    for (int dd = 0; dd < Dn; ++dd)
      a += ca[((size_t)(b0 + m) * Tm1) * Dn + dd] * iw0[w * Dn + dd];
    Fb[i] = fmaxf(a, 0.f);
  }
  __syncthreads();
  for (int i = tid; i < MB * Wn; i += 256) {           // h1 = relu(h0 @ iw1.T + ib1)
    int m = i >> 7, w = i & 127;
    float a = ib1[w];
    const float* hr = Fb + m * Wn;
    for (int k = 0; k < Wn; ++k) a += hr[k] * iw1[w * Wn + k];
    Fb[MB * Wn + i] = fmaxf(a, 0.f);
  }
  __syncthreads();
  for (int i = tid; i < MB * Hn; i += 256) {           // y0 = h1 @ iw2.T + ib2
    int m = i >> 6, hh = i & 63;
    float a = ib2[hh];
    const float* hr = Fb + MB * Wn + m * Wn;
    for (int k = 0; k < Wn; ++k) a += hr[k] * iw2[hh * Wn + k];
    yf[i] = a;
    ysb[i] = f2bf(a);
    ksum[i] = 0.f;
  }
  __syncthreads();

  // ---- scan over 1023 RK4 steps ----
#pragma unroll 1
  for (int t = 0; t < Tm1; ++t) {
    float hstep = ts[t + 1] - ts[t];
    if (tid < 128) {   // waves 0..3, fully enabled (EXEC all-ones per wave)
      int row = tid >> 3, dd = tid & 7;
      size_t gi = ((size_t)(b0 + row) * Tm1 + t) * Dn + dd;
      unsigned goff = (unsigned)(gi * sizeof(float));   // per-lane byte offset (GVS)
      unsigned l_d  = (unsigned)(uintptr_t)(dCs + tid); // LDS byte address = addr[31:0]
      unsigned l_c  = (unsigned)(uintptr_t)(cCs + tid);
      unsigned l_b  = (unsigned)(uintptr_t)(bCs + tid);
      // CDNA5 async copy global->LDS, tracked with ASYNCcnt (no VGPR data path)
      asm volatile(
          "global_load_async_to_lds_b32 %0, %3, %4\n\t"
          "global_load_async_to_lds_b32 %1, %3, %5\n\t"
          "global_load_async_to_lds_b32 %2, %3, %6\n\t"
          "s_wait_asynccnt 0x0"
          :: "v"(l_d), "v"(l_c), "v"(l_b), "v"(goff),
             "s"(cd), "s"(cc), "s"(cb)
          : "memory");
      if (t + 8 < Tm1) {                               // gfx1250 global_prefetch_b8
        __builtin_prefetch(cd + gi + 64, 0, 1);
        __builtin_prefetch(cc + gi + 64, 0, 1);
        __builtin_prefetch(cb + gi + 64, 0, 1);
      }
    }
    __syncthreads();

#pragma unroll
    for (int st = 0; st < 4; ++st) {
      const float sfac = (st == 0) ? 0.f : ((st == 3) ? 1.f : 0.5f);
      const float alph = (st == 2) ? 1.f : 0.5f;
      const float wgt  = (st == 1 || st == 2) ? 2.f : 1.f;

      if (tid < 128) {                                 // dX = (3ds+2c)s + b
        float s = sfac * hstep;
        dxs[tid] = (3.f * dCs[tid] * s + 2.f * cCs[tid]) * s + bCs[tid];
      }
      // layer 1: [16,64] x [64,128] -> softplus -> h1b   (8 waves x 1 tile)
      {
        v8f acc = {};
        acc = wave_gemm_tile<Hn>(ysb, fw0u, wave, lane, acc);
        int n = wave * 16 + (lane & 15);
        float bs = fb0s[n];
#pragma unroll
        for (int r = 0; r < 8; ++r) {
          int m = r + 8 * (lane >> 4);
          h1b[m * Wn + n] = f2bf(softplus_f(acc[r] + bs));
        }
      }
      __syncthreads();
      // layer 2: [16,128] x [128,128] -> softplus -> h2b
      {
        v8f acc = {};
        acc = wave_gemm_tile<Wn>(h1b, fw1u, wave, lane, acc);
        int n = wave * 16 + (lane & 15);
        float bs = fb1s[n];
#pragma unroll
        for (int r = 0; r < 8; ++r) {
          int m = r + 8 * (lane >> 4);
          h2b[m * Wn + n] = f2bf(softplus_f(acc[r] + bs));
        }
      }
      __syncthreads();
      // layer 3: [16,128] x [128,512] -> tanh -> Fb   (8 waves x 4 tiles)
#pragma unroll
      for (int tt = 0; tt < 4; ++tt) {
        int nt = wave + tt * 8;
        v8f acc = {};
        acc = wave_gemm_tile<Wn>(h2b, fw2u, nt, lane, acc);
        int n = nt * 16 + (lane & 15);
        float bs = fb2s[n];
#pragma unroll
        for (int r = 0; r < 8; ++r) {
          int m = r + 8 * (lane >> 4);
          Fb[m * HDn + n] = tanh_f(acc[r] + bs);
        }
      }
      __syncthreads();
      // einsum over D=8:  k[m,h] = sum_d F[m,h,d] * dX[m,d]
      for (int i = tid; i < MB * Hn; i += 256) {
        int m = i >> 6, hh = i & 63;
        const float* fr = Fb + m * HDn + hh * Dn;
        const float* dx = dxs + m * Dn;
        float s = 0.f;
#pragma unroll
        for (int dd = 0; dd < Dn; ++dd) s += fr[dd] * dx[dd];
        kst[i] = s;
      }
      __syncthreads();
      // RK accumulate + next stage input
      for (int i = tid; i < MB * Hn; i += 256) {
        float kv = kst[i];
        ksum[i] += wgt * kv;
        if (st < 3) ysb[i] = f2bf(yf[i] + alph * hstep * kv);
      }
      __syncthreads();
    }

    // y += (h/6) * (k1 + 2k2 + 2k3 + k4)
    for (int i = tid; i < MB * Hn; i += 256) {
      float yn = yf[i] + (hstep * (1.f / 6.f)) * ksum[i];
      yf[i] = yn;
      ysb[i] = f2bf(yn);
      ksum[i] = 0.f;
    }
    __syncthreads();
  }

  // ---- readout: sigmoid(y @ lw.T + lb) ----
  if (tid < MB) {
    float a = lb[0];
    for (int hh = 0; hh < Hn; ++hh) a += yf[tid * Hn + hh] * lw[hh];
    out[b0 + tid] = 1.f / (1.f + __expf(-a));
  }
}

extern "C" void kernel_launch(void* const* d_in, const int* in_sizes, int n_in,
                              void* d_out, int out_size, void* d_ws, size_t ws_size,
                              hipStream_t stream) {
  (void)in_sizes; (void)n_in; (void)d_ws; (void)ws_size; (void)out_size;
  const float* ts  = (const float*)d_in[0];
  const float* cd  = (const float*)d_in[1];
  const float* cc  = (const float*)d_in[2];
  const float* cb  = (const float*)d_in[3];
  const float* ca  = (const float*)d_in[4];
  const float* iw0 = (const float*)d_in[5];
  const float* ib0 = (const float*)d_in[6];
  const float* iw1 = (const float*)d_in[7];
  const float* ib1 = (const float*)d_in[8];
  const float* iw2 = (const float*)d_in[9];
  const float* ib2 = (const float*)d_in[10];
  const float* fw0 = (const float*)d_in[11];
  const float* fb0 = (const float*)d_in[12];
  const float* fw1 = (const float*)d_in[13];
  const float* fb1 = (const float*)d_in[14];
  const float* fw2 = (const float*)d_in[15];
  const float* fb2 = (const float*)d_in[16];
  const float* lw  = (const float*)d_in[17];
  const float* lb  = (const float*)d_in[18];
  float* out = (float*)d_out;

  // packed weights + biases + state/activation staging (fits in 320KB WGP LDS)
  constexpr size_t LDS_BYTES =
      (4096u + 8192u + 32768u) * 4        // bf16 fragment-packed weights
    + (128u + 128u + 512u) * 4            // biases
    + (3u * 1024u + 8192u) * 4            // yf, ksum, kst, Fb
    + (4u * 128u) * 4                     // dCs, cCs, bCs, dxs
    + (1024u + 2048u + 2048u) * 2;        // ysb, h1b, h2b (bf16)

  ncde_kernel<<<dim3(8), dim3(256), LDS_BYTES, stream>>>(
      ts, cd, cc, cb, ca, iw0, ib0, iw1, ib1, iw2, ib2,
      fw0, fb0, fw1, fb1, fw2, fb2, lw, lb, out);
}
